// BiLSTMClassifier_21088289423525
// MI455X (gfx1250) — compile-verified, matrix operand order checked by reading, not run on hardware
//
#include <hip/hip_runtime.h>
#include <hip/hip_bf16.h>
#include <hip/hip_fp16.h>

typedef __attribute__((ext_vector_type(16))) _Float16 v16h;
typedef __attribute__((ext_vector_type(8)))  _Float16 v8h;
typedef __attribute__((ext_vector_type(8)))  float    v8f;
typedef int v4i_t __attribute__((vector_size(16)));   // matches async-LDS builtin param

#define EE   300
#define EP   320            // E padded to multiple of 32
#define HH   256            // hidden
#define G4   1024           // 4H
#define CC   3
#define BB   64
#define TT   256
#define MM   (BB*TT)        // 16384 rows of x
#define NSPLIT 4            // workgroups per direction in the recurrence
#define GS   264            // LDS gate-row stride (floats), 256 + 8 pad
#define WLS  264            // LDS W_hh-slice row stride (halves), 256 + 8 pad

#if __has_builtin(__builtin_amdgcn_global_load_async_to_lds_b128) && \
    __has_builtin(__builtin_amdgcn_s_wait_asynccnt)
#define USE_ASYNC_LDS 1
#else
#define USE_ASYNC_LDS 0
#endif

// ---------------------------------------------------------------------------
// WMMA fragment helpers (CDNA5 16-bit A/B layout, ISA 7.12.2)
// A-frag and B-frag have the same per-lane pattern when the source matrix is
// stored "outer-dim-major, K contiguous" (A: [M][K] row-major, B: [N][K]
// row-major, i.e. W stored as [out_features][in_features] == native layout).
// lane (0..15): rows 0..15, K chunks {0..7, 16..23}
// lane (16..31): rows 0..15, K chunks {8..15, 24..31}
// ---------------------------------------------------------------------------
__device__ __forceinline__ v16h load_ab_frag(const _Float16* __restrict__ tile, int ld) {
  int lane = threadIdx.x & 31;
  const _Float16* p = tile + (lane & 15) * (size_t)ld + ((lane & 16) ? 8 : 0);
  v8h lo = *reinterpret_cast<const v8h*>(p);
  v8h hi = *reinterpret_cast<const v8h*>(p + 16);
  v16h f;
#pragma unroll
  for (int i = 0; i < 8; ++i) { f[i] = lo[i]; f[i + 8] = hi[i]; }
  return f;
}

__device__ __forceinline__ float sigmoidf_(float x) { return 1.0f / (1.0f + __expf(-x)); }

// ---------------------------------------------------------------------------
// Prep: f32 -> f16 with column padding (handles W_ih pad 300->320, W_hh copy)
// ---------------------------------------------------------------------------
__global__ void cvt_pad_kernel(const float* __restrict__ src, _Float16* __restrict__ dst,
                               int rows, int cin, int cout) {
  int i = blockIdx.x * blockDim.x + threadIdx.x;
  if (i >= rows * cout) return;
  int r = i / cout, c = i - r * cout;
  dst[i] = (_Float16)((c < cin) ? src[(size_t)r * cin + c] : 0.0f);
}

__global__ void bias_add_kernel(const float* __restrict__ a, const float* __restrict__ b,
                                float* __restrict__ o, int n) {
  int i = blockIdx.x * blockDim.x + threadIdx.x;
  if (i < n) o[i] = a[i] + b[i];
}

// Embedding gather -> f16, padded to EP columns. x row m = b*T + t.
__global__ void gather_kernel(const int* __restrict__ ids, const float* __restrict__ emb,
                              _Float16* __restrict__ x16) {
  int i = blockIdx.x * blockDim.x + threadIdx.x;
  if (i >= MM * EP) return;
  int m = i / EP, col = i - m * EP;
  float v = 0.0f;
  if (col < EE) v = emb[(size_t)ids[m] * EE + col];
  x16[i] = (_Float16)v;
}

// Zero h0 double-buffers and the cross-WG barrier counters.
__global__ void init_kernel(_Float16* __restrict__ hbuf, unsigned int* __restrict__ bar) {
  int i = blockIdx.x * blockDim.x + threadIdx.x;
  if (i < 2 * 2 * BB * HH) hbuf[i] = (_Float16)0.0f;
  if (i < 2) bar[i] = 0u;
}

// ---------------------------------------------------------------------------
// pre[dir][t][b][4H] = x @ W_ih^T + (b_ih + b_hh)   (f16 output, f32 accum)
// grid: (MM/64, G4/128, 2), 256 threads (8 waves). Each wave: 4 16x16 tiles.
// ---------------------------------------------------------------------------
__global__ void __launch_bounds__(256)
gemm_pre_kernel(const _Float16* __restrict__ x16,
                const _Float16* __restrict__ wih16,   // [2][G4*EP]
                const float*   __restrict__ biasc,    // [2][G4]
                _Float16*      __restrict__ pre) {    // [2][TT*BB*G4]
  int mb = blockIdx.x, nb = blockIdx.y, dir = blockIdx.z;
  const _Float16* W = wih16 + (size_t)dir * G4 * EP;
  const float* bias = biasc + dir * G4;
  _Float16* out = pre + (size_t)dir * TT * BB * G4;

  int w = threadIdx.x >> 5, lane = threadIdx.x & 31;
  int mtile = w & 3, nh = w >> 2;
  int mrow0 = mb * 64 + mtile * 16;

  v8f acc[4] = {};
  for (int k0 = 0; k0 < EP; k0 += 32) {
    v16h a = load_ab_frag(x16 + (size_t)mrow0 * EP + k0, EP);
#pragma unroll
    for (int j = 0; j < 4; ++j) {
      int ng = nb * 128 + (nh * 4 + j) * 16;
      v16h b = load_ab_frag(W + (size_t)ng * EP + k0, EP);
      acc[j] = __builtin_amdgcn_wmma_f32_16x16x32_f16(false, a, false, b,
                                                      (short)0, acc[j], false, false);
    }
  }
  int n = lane & 15, mo = (lane & 16) ? 8 : 0;
#pragma unroll
  for (int j = 0; j < 4; ++j) {
    int ng = nb * 128 + (nh * 4 + j) * 16 + n;
    float bv = bias[ng];
#pragma unroll
    for (int r = 0; r < 8; ++r) {
      int m = mrow0 + mo + r;            // m = b*T + t
      int bi = m >> 8, t = m & 255;      // T == 256
      out[((size_t)t * BB + bi) * G4 + ng] = (_Float16)(acc[j][r] + bv);
    }
  }
}

// ---------------------------------------------------------------------------
// Persistent recurrence. grid = 8 blocks: dir = blk/4, hidden-slice j = blk%4.
// W_hh slice (256x256 f16 = 128KB) is staged in LDS once via async
// global->LDS copies (ASYNCcnt), so the per-step GEMM critical path reads B
// fragments with ds_load_b128 instead of L2 round-trips. Gates live in LDS,
// c-state in registers, h-slice exchanged via L2 + atomic barrier.
// ---------------------------------------------------------------------------
__global__ void __launch_bounds__(256)
lstm_rec_kernel(const _Float16* __restrict__ whh16,  // [2][G4*HH]
                const _Float16* __restrict__ pre,    // [2][TT*BB*G4]
                _Float16*       __restrict__ hs,     // [2][BB*TT*HH]
                _Float16*       __restrict__ hbuf,   // [2][2][BB*HH]
                unsigned int*   __restrict__ bar) {  // [2]
  extern __shared__ char smem[];
  float* g = reinterpret_cast<float*>(smem);                    // [64][GS]
  _Float16* Wl = reinterpret_cast<_Float16*>(smem + 64 * GS * 4); // [256][WLS]

  int dir = blockIdx.x >> 2, jsl = blockIdx.x & 3;
  const _Float16* W = whh16 + (size_t)dir * G4 * HH;
  const _Float16* P = pre + (size_t)dir * TT * BB * G4;
  _Float16* HS = hs + (size_t)dir * BB * TT * HH;
  _Float16* HB = hbuf + (size_t)dir * 2 * BB * HH;
  unsigned int* barp = bar + dir;

  int tid = threadIdx.x;
  int w = tid >> 5, lane = tid & 31;
  int mtile = w & 3, nh = w >> 2;
  int n = lane & 15, mo = (lane & 16) ? 8 : 0;

  // ---- stage this WG's W_hh slice into LDS (local row ln = q*64 + r) ----
  // 256 rows x 256 halves = 8192 16-byte chunks, 32 per thread.
  for (int idx = tid; idx < 256 * 32; idx += 256) {
    int ln = idx >> 5, ck = idx & 31;
    int Ng = ((ln >> 6) << 8) + jsl * 64 + (ln & 63);
    const _Float16* src = W + (size_t)Ng * HH + ck * 8;
    _Float16* dst = Wl + (size_t)ln * WLS + ck * 8;
#if USE_ASYNC_LDS
    __builtin_amdgcn_global_load_async_to_lds_b128(
        (__attribute__((address_space(1))) v4i_t*)src,
        (__attribute__((address_space(3))) v4i_t*)dst, 0, 0);
#else
    *reinterpret_cast<uint4*>(dst) = *reinterpret_cast<const uint4*>(src);
#endif
  }
#if USE_ASYNC_LDS
  __builtin_amdgcn_s_wait_asynccnt(0);
#endif
  __syncthreads();

  int brow = tid >> 2;                 // batch row this thread owns (activation)
  int colb = (tid & 3) * 16;           // 16 contiguous slice-columns
  float creg[16];
#pragma unroll
  for (int r = 0; r < 16; ++r) creg[r] = 0.0f;

  for (int t = 0; t < TT; ++t) {
    int tt = dir ? (TT - 1 - t) : t;
    const _Float16* hc = HB + (size_t)(t & 1) * BB * HH;
    _Float16* hn = HB + (size_t)((t + 1) & 1) * BB * HH;

    v8f acc[8];
#pragma unroll
    for (int i = 0; i < 8; ++i) {      // seed accumulators from pre (has bias)
      int nt = nh * 8 + i, q = nt >> 2;
      int Ng = q * 256 + jsl * 64 + (nt & 3) * 16;
      const _Float16* cp = P + ((size_t)tt * BB + mtile * 16) * G4 + Ng;
#pragma unroll
      for (int r = 0; r < 8; ++r) acc[i][r] = (float)cp[(size_t)(mo + r) * G4 + n];
    }
    // prefetch next step's pre fragments (HBM stream) -> global_prefetch_b8
    if (t + 1 < TT) {
      int tn = dir ? (TT - 2 - t) : (t + 1);
#pragma unroll
      for (int i = 0; i < 8; ++i) {
        int nt = nh * 8 + i, q = nt >> 2;
        int Ng = q * 256 + jsl * 64 + (nt & 3) * 16;
        __builtin_prefetch(P + ((size_t)tn * BB + mtile * 16 + mo + i) * G4 + Ng + n, 0, 1);
      }
    }
#pragma unroll
    for (int k = 0; k < 8; ++k) {
      v16h a = load_ab_frag(hc + (size_t)(mtile * 16) * HH + k * 32, HH);
#pragma unroll
      for (int i = 0; i < 8; ++i) {
        int nt = nh * 8 + i;
        // B fragment from LDS-resident W slice (local row = nt*16 + lane&15)
        v16h bf = load_ab_frag(Wl + (size_t)(nt * 16) * WLS + k * 32, WLS);
        acc[i] = __builtin_amdgcn_wmma_f32_16x16x32_f16(false, a, false, bf,
                                                        (short)0, acc[i], false, false);
      }
    }
    // gates -> LDS (local col = nt*16 + n == q*64 + within-gate col)
#pragma unroll
    for (int i = 0; i < 8; ++i) {
      int nt = nh * 8 + i;
#pragma unroll
      for (int r = 0; r < 8; ++r)
        g[(mtile * 16 + mo + r) * GS + nt * 16 + n] = acc[i][r];
    }
    __syncthreads();

    // activation: i,f,g,o at local offsets 0/64/128/192
#pragma unroll
    for (int r = 0; r < 16; ++r) {
      int col = colb + r;
      float xi = g[brow * GS + col];
      float xf = g[brow * GS + 64 + col];
      float xg = g[brow * GS + 128 + col];
      float xo = g[brow * GS + 192 + col];
      float cv = sigmoidf_(xf) * creg[r] + sigmoidf_(xi) * tanhf(xg);
      creg[r] = cv;
      float hv = sigmoidf_(xo) * tanhf(cv);
      int gcol = jsl * 64 + col;
      hn[brow * HH + gcol] = (_Float16)hv;
      HS[((size_t)brow * TT + tt) * HH + gcol] = (_Float16)hv;
    }
    __threadfence();                   // make hn visible device-wide
    __syncthreads();
    if (tid == 0) {                    // cross-WG arrive + spin (monotonic)
      __hip_atomic_fetch_add(barp, 1u, __ATOMIC_RELEASE, __HIP_MEMORY_SCOPE_AGENT);
      unsigned int target = (unsigned)(NSPLIT * (t + 1));
      while (__hip_atomic_load(barp, __ATOMIC_ACQUIRE, __HIP_MEMORY_SCOPE_AGENT) < target)
        __builtin_amdgcn_s_sleep(2);
    }
    __syncthreads();
  }
}

// ---------------------------------------------------------------------------
// Masked mean-pool over T then logits = pooled @ W_c^T + b_c. One block per b.
// ---------------------------------------------------------------------------
__global__ void __launch_bounds__(512)
pool_cls_kernel(const int* __restrict__ mask, const _Float16* __restrict__ hs,
                const float* __restrict__ Wc, const float* __restrict__ bc,
                float* __restrict__ out) {
  __shared__ float pooled[2 * HH];
  int b = blockIdx.x, nn = threadIdx.x;
  const _Float16* base = hs + (nn < HH ? (size_t)0 : (size_t)BB * TT * HH)
                         + (size_t)b * TT * HH + (nn & (HH - 1));
  float s = 0.0f, msum = 0.0f;
  for (int t = 0; t < TT; ++t) {
    int mk = mask[b * TT + t];
    msum += (float)mk;
    if (mk) s += (float)base[(size_t)t * HH];
  }
  pooled[nn] = s / fmaxf(msum, 1e-9f);
  __syncthreads();
  if (nn < CC) {
    float acc = bc[nn];
    for (int k = 0; k < 2 * HH; ++k) acc += pooled[k] * Wc[nn * 2 * HH + k];
    out[b * CC + nn] = acc;
  }
}

// ---------------------------------------------------------------------------
extern "C" void kernel_launch(void* const* d_in, const int* in_sizes, int n_in,
                              void* d_out, int out_size, void* d_ws, size_t ws_size,
                              hipStream_t stream) {
  const int*   ids    = (const int*)d_in[0];
  const int*   amask  = (const int*)d_in[1];
  const float* emb    = (const float*)d_in[2];
  const float* Wih_f  = (const float*)d_in[3];
  const float* Whh_f  = (const float*)d_in[4];
  const float* bih_f  = (const float*)d_in[5];
  const float* bhh_f  = (const float*)d_in[6];
  const float* Wih_b  = (const float*)d_in[7];
  const float* Whh_b  = (const float*)d_in[8];
  const float* bih_b  = (const float*)d_in[9];
  const float* bhh_b  = (const float*)d_in[10];
  const float* Wc     = (const float*)d_in[11];
  const float* bc     = (const float*)d_in[12];
  float* out = (float*)d_out;

  // workspace carve-up (256B aligned)
  char* base = (char*)d_ws;
  size_t off = 0;
  auto take = [&](size_t bytes) { char* p = base + off; off = (off + bytes + 255) & ~(size_t)255; return p; };
  _Float16* x16   = (_Float16*)take((size_t)MM * EP * 2);
  _Float16* wih16 = (_Float16*)take((size_t)2 * G4 * EP * 2);
  _Float16* whh16 = (_Float16*)take((size_t)2 * G4 * HH * 2);
  float*    biasc = (float*)   take((size_t)2 * G4 * 4);
  _Float16* pre   = (_Float16*)take((size_t)2 * TT * BB * G4 * 2);
  _Float16* hsbuf = (_Float16*)take((size_t)2 * BB * TT * HH * 2);
  _Float16* hbuf  = (_Float16*)take((size_t)2 * 2 * BB * HH * 2);
  unsigned int* bar = (unsigned int*)take(256);
  (void)in_sizes; (void)n_in; (void)out_size; (void)ws_size;

  const int TB = 256;
  // weight conversion (f32 -> f16, K padded for W_ih)
  cvt_pad_kernel<<<(G4 * EP + TB - 1) / TB, TB, 0, stream>>>(Wih_f, wih16,             G4, EE, EP);
  cvt_pad_kernel<<<(G4 * EP + TB - 1) / TB, TB, 0, stream>>>(Wih_b, wih16 + (size_t)G4 * EP, G4, EE, EP);
  cvt_pad_kernel<<<(G4 * HH + TB - 1) / TB, TB, 0, stream>>>(Whh_f, whh16,             G4, HH, HH);
  cvt_pad_kernel<<<(G4 * HH + TB - 1) / TB, TB, 0, stream>>>(Whh_b, whh16 + (size_t)G4 * HH, G4, HH, HH);
  bias_add_kernel<<<(G4 + TB - 1) / TB, TB, 0, stream>>>(bih_f, bhh_f, biasc,      G4);
  bias_add_kernel<<<(G4 + TB - 1) / TB, TB, 0, stream>>>(bih_b, bhh_b, biasc + G4, G4);
  // embedding gather
  gather_kernel<<<((size_t)MM * EP + TB - 1) / TB, TB, 0, stream>>>(ids, emb, x16);
  // zero h0 + barrier counters
  init_kernel<<<(2 * 2 * BB * HH + TB - 1) / TB, TB, 0, stream>>>(hbuf, bar);
  // big input-projection GEMM (WMMA)
  gemm_pre_kernel<<<dim3(MM / 64, G4 / 128, 2), 256, 0, stream>>>(x16, wih16, biasc, pre);
  // persistent recurrence: gates (64*GS*4 = 67584B) + W slice (256*WLS*2 = 135168B)
  const int shmem = 64 * GS * 4 + 256 * WLS * 2;   // 202752 bytes < 320KB WGP LDS
  (void)hipFuncSetAttribute(reinterpret_cast<const void*>(lstm_rec_kernel),
                            hipFuncAttributeMaxDynamicSharedMemorySize, shmem);
  lstm_rec_kernel<<<2 * NSPLIT, 256, shmem, stream>>>(whh16, pre, hsbuf, hbuf, bar);
  // masked mean pool + classifier
  pool_cls_kernel<<<BB, 512, 0, stream>>>(amask, hsbuf, Wc, bc, out);
}
